// SwinAttention_19834158973060
// MI455X (gfx1250) — matrix operand order repaired and checked by reading, not compiled, not run on hardware
//
#include <hip/hip_runtime.h>
#include <hip/hip_bf16.h>

// ---------------------------------------------------------------------------
// Swin window attention for MI455X (gfx1250, wave32, WMMA) — transposed form.
//
// 16384 windows (B*H*NW), 49x32 Q/K/V each. Memory floor ~18us at 23.3 TB/s;
// the kernel is kept instruction-lean so issue rate doesn't dominate:
//   * S^T = K Q^T so one query's 64 scores live in 2 lanes -> xor-16 lane
//     swap (v_permlanex16_b32, single VALU op) per softmax reduction,
//   * raw v_exp_f32 (__builtin_amdgcn_exp2f) for softmax: s-m <= 0 and
//     flush-to-zero underflow is exactly the desired masked/tiny behavior,
//   * P^T assembled in registers as WMMA B-fragments (v_cvt_pk_bf16_f32 +
//     permlanex16), no P LDS round-trip,
//   * O^T fragments hold 8 consecutive head dims per lane -> float4 stores,
//   * 1/sqrt(d)*log2(e) folded into Q staging, 1/rowsum folded into store.
// ---------------------------------------------------------------------------

typedef __attribute__((ext_vector_type(16))) __bf16 v16bf;
typedef __attribute__((ext_vector_type(8)))  float  v8f;

#define BB   8
#define HN   8
#define HI   112
#define WIW  112
#define MW   7
#define SS   3
#define DHD  32
#define NTOK (HI * WIW)           // 12544
#define NWIN 256
// 1/sqrt(32) * log2(e): scores computed directly in exp2 domain.
#define QS2  (0.17677669529663687f * 1.4426950408889634f)

// ---- lane xor-16 swap: one v_permlanex16_b32 on wave32 ----
__device__ __forceinline__ int xor16_i(int x) {
#if __has_builtin(__builtin_amdgcn_permlanex16)
    return __builtin_amdgcn_permlanex16(x, x, 0x76543210, 0xfedcba98,
                                        false, false);
#else
    return __shfl_xor(x, 16);
#endif
}
__device__ __forceinline__ float xor16_f(float x) {
    return __builtin_bit_cast(float, xor16_i(__builtin_bit_cast(int, x)));
}
__device__ __forceinline__ unsigned xor16_u(unsigned x) {
    return (unsigned)xor16_i((int)x);
}

// ---- raw hardware exp2 (v_exp_f32); underflow flushes to 0 (desired) ----
__device__ __forceinline__ float exp2_raw(float x) {
#if __has_builtin(__builtin_amdgcn_exp2f)
    return __builtin_amdgcn_exp2f(x);
#else
    return __builtin_exp2f(x);
#endif
}

// Assemble a 16-element bf16 WMMA fragment from two 16-byte LDS chunks.
__device__ __forceinline__ v16bf frag2(const __bf16* p0, const __bf16* p1) {
    union { uint4 u[2]; v16bf v; } f;
    f.u[0] = *reinterpret_cast<const uint4*>(p0);
    f.u[1] = *reinterpret_cast<const uint4*>(p1);
    return f.v;
}

// Pack two f32 into one dword of 2 bf16 (lo,hi) -> v_cvt_pk_bf16_f32.
__device__ __forceinline__ unsigned pkbf(float lo, float hi) {
    union { __bf16 b[2]; unsigned u; } t;
    t.b[0] = (__bf16)lo; t.b[1] = (__bf16)hi;
    return t.u;
}

__launch_bounds__(128)
__global__ void swin_attn_wmma(const float* __restrict__ Qg,
                               const float* __restrict__ Kg,
                               const float* __restrict__ Vg,
                               float* __restrict__ Og) {
    // LDS staging, 49 tokens padded to 64 (pad rows zeroed).
    // sQ is pre-scaled by QS2. sVt is V transposed (dim-major) so that both
    // GEMMs read contiguous LDS chunks for their fragments.
    __shared__ alignas(16) __bf16 sQ[64 * 32];
    __shared__ alignas(16) __bf16 sK[64 * 32];
    __shared__ alignas(16) __bf16 sVt[32 * 64];

    const int blk = blockIdx.x;
    const int w   = blk & (NWIN - 1);
    const int bh  = blk >> 8;
    const int wy  = w >> 4;
    const int wx  = w & 15;
    const size_t base = (size_t)bh * (size_t)NTOK * DHD;

    const int tid = threadIdx.x;

    // ---- Stage window Q/K/V: gather through cyclic shift, fp32 -> bf16 ----
    for (int idx = tid; idx < 512; idx += 128) {   // 64 tokens * 8 float4
        const int t = idx >> 3;
        const int q = idx & 7;
        float4 vq = make_float4(0.f, 0.f, 0.f, 0.f);
        float4 vk = vq, vv = vq;
        if (t < 49) {
            const int i = t / 7, j = t - i * 7;
            int r = wy * MW + i + SS; if (r >= HI)  r -= HI;
            int c = wx * MW + j + SS; if (c >= WIW) c -= WIW;
            const size_t off = base + (size_t)(r * WIW + c) * DHD + q * 4;
            vq = *reinterpret_cast<const float4*>(Qg + off);
            vk = *reinterpret_cast<const float4*>(Kg + off);
            vv = *reinterpret_cast<const float4*>(Vg + off);
        }
        const int o = t * 32 + q * 4;
        { union { __bf16 b[4]; uint2 u; } tq;
          tq.b[0] = (__bf16)(vq.x * QS2); tq.b[1] = (__bf16)(vq.y * QS2);
          tq.b[2] = (__bf16)(vq.z * QS2); tq.b[3] = (__bf16)(vq.w * QS2);
          *reinterpret_cast<uint2*>(&sQ[o]) = tq.u; }
        { union { __bf16 b[4]; uint2 u; } tk;
          tk.b[0] = (__bf16)vk.x; tk.b[1] = (__bf16)vk.y;
          tk.b[2] = (__bf16)vk.z; tk.b[3] = (__bf16)vk.w;
          *reinterpret_cast<uint2*>(&sK[o]) = tk.u; }
        const int d0 = q * 4;
        sVt[(d0 + 0) * 64 + t] = (__bf16)vv.x;
        sVt[(d0 + 1) * 64 + t] = (__bf16)vv.y;
        sVt[(d0 + 2) * 64 + t] = (__bf16)vv.z;
        sVt[(d0 + 3) * 64 + t] = (__bf16)vv.w;
    }
    __syncthreads();

    const int wv   = tid >> 5;       // wave id: owns query tile wv (16 queries)
    const int lane = tid & 31;
    const int h    = lane >> 4;      // lane half
    const int lrow = lane & 15;

    // ---- S^T = K Q^T : A = K tile (16 keys x 32), B = Q^T (32 x 16 q) ----
    const int qrow = wv * 16 + lrow;
    const v16bf bQ = frag2(&sQ[qrow * 32 + h * 8],
                           &sQ[qrow * 32 + 16 + h * 8]);

    v8f st[4];
#pragma unroll
    for (int tt = 0; tt < 4; ++tt) {
        const int trow = tt * 16 + lrow;     // A frag row = key index
        const v16bf aK = frag2(&sK[trow * 32 + h * 8],
                               &sK[trow * 32 + 16 + h * 8]);
        v8f z = {0.f, 0.f, 0.f, 0.f, 0.f, 0.f, 0.f, 0.f};
        st[tt] = __builtin_amdgcn_wmma_f32_16x16x32_bf16(
            false, aK, false, bQ, (short)0, z, false, false);
    }
    // Lane holds S^T(key t = 16*tt + 8*h + r, query q = wv*16 + lrow):
    // one query's 64 scores live in exactly lanes {lrow, lrow+16}.

    // ---- softmax (exp2 domain; keys >= 49 masked by construction) ----
    // valid keys: tiles 0..2 fully (t <= 47); tile 3 only (h==0, r==0) t=48.
    float m = (h == 0) ? st[3][0] : -3.0e38f;
#pragma unroll
    for (int r = 0; r < 8; ++r)
        m = fmaxf(m, fmaxf(fmaxf(st[0][r], st[1][r]), st[2][r]));
    m = fmaxf(m, xor16_f(m));                 // combine the two lane-halves

    float e0[8], e1[8], e2[8];
#pragma unroll
    for (int r = 0; r < 8; ++r) {
        e0[r] = exp2_raw(st[0][r] - m);
        e1[r] = exp2_raw(st[1][r] - m);
        e2[r] = exp2_raw(st[2][r] - m);
    }
    float e3 = exp2_raw(st[3][0] - m);
    e3 = (h == 0) ? e3 : 0.f;                 // key 48 valid only in half 0

    float rs = e3;
#pragma unroll
    for (int r = 0; r < 8; ++r) rs += e0[r] + e1[r] + e2[r];
    rs += xor16_f(rs);
    const float rinv = __builtin_amdgcn_rcpf(rs);   // folded into O store

    // ---- Build P^T WMMA B-fragments in registers (no LDS round-trip) ----
    // B frag (K-step ks covers keys 32ks..32ks+31): lane needs keys
    // 32ks+16h+0..15 for its query column; own regs hold 8 of them, the
    // xor-16 partner lane holds the other 8.
    unsigned p00[4], p01[4], p10[4], p11[4];
#pragma unroll
    for (int jj = 0; jj < 4; ++jj) {
        p00[jj] = pkbf(e0[2 * jj], e0[2 * jj + 1]);   // tile 0
        p01[jj] = pkbf(e1[2 * jj], e1[2 * jj + 1]);   // tile 1
        p10[jj] = pkbf(e2[2 * jj], e2[2 * jj + 1]);   // tile 2
        p11[jj] = 0u;                                  // tile 3: zeros
    }
    p11[0] = pkbf(e3, 0.f);                            // key 48

    union { unsigned u[8]; v16bf v; } bP0, bP1;
#pragma unroll
    for (int jj = 0; jj < 4; ++jj) {
        const unsigned x00 = xor16_u(p00[jj]);
        const unsigned x01 = xor16_u(p01[jj]);
        const unsigned x10 = xor16_u(p10[jj]);
        const unsigned x11 = xor16_u(p11[jj]);
        bP0.u[jj]     = h ? x01     : p00[jj];   // ks=0, K 0-7
        bP0.u[4 + jj] = h ? p01[jj] : x00;       // ks=0, K 8-15
        bP1.u[jj]     = h ? x11     : p10[jj];   // ks=1, K 0-7
        bP1.u[4 + jj] = h ? p11[jj] : x10;       // ks=1, K 8-15
    }

    // ---- O^T = V^T P^T : A = V^T tile (16 dims x 32 keys) from sVt ----
    v8f oacc[2];
#pragma unroll
    for (int ta = 0; ta < 2; ++ta) {
        const int drow = ta * 16 + lrow;             // A frag row = head dim
        const v16bf aV0 = frag2(&sVt[drow * 64 + h * 16],
                                &sVt[drow * 64 + h * 16 + 8]);
        const v16bf aV1 = frag2(&sVt[drow * 64 + 32 + h * 16],
                                &sVt[drow * 64 + 32 + h * 16 + 8]);
        v8f z = {0.f, 0.f, 0.f, 0.f, 0.f, 0.f, 0.f, 0.f};
        z = __builtin_amdgcn_wmma_f32_16x16x32_bf16(
            false, aV0, false, bP0.v, (short)0, z, false, false);
        oacc[ta] = __builtin_amdgcn_wmma_f32_16x16x32_bf16(
            false, aV1, false, bP1.v, (short)0, z, false, false);
    }
    // Lane holds O^T(dim d = 16ta + 8h + r, query q): 8 consecutive dims.

    // ---- Store O (shifted coordinates, like reference), vectorized ----
    const int q = wv * 16 + lrow;
    if (q < 49) {
        const int i = q / 7, j = q - i * 7;
        const int n_out = (wy * MW + i) * WIW + (wx * MW + j);
        float* op = Og + base + (size_t)n_out * DHD;
#pragma unroll
        for (int ta = 0; ta < 2; ++ta) {
            const int d0 = ta * 16 + 8 * h;
            float4 lo4 = make_float4(oacc[ta][0] * rinv, oacc[ta][1] * rinv,
                                     oacc[ta][2] * rinv, oacc[ta][3] * rinv);
            float4 hi4 = make_float4(oacc[ta][4] * rinv, oacc[ta][5] * rinv,
                                     oacc[ta][6] * rinv, oacc[ta][7] * rinv);
            *reinterpret_cast<float4*>(op + d0)     = lo4;
            *reinterpret_cast<float4*>(op + d0 + 4) = hi4;
        }
    }
}

extern "C" void kernel_launch(void* const* d_in, const int* in_sizes, int n_in,
                              void* d_out, int out_size, void* d_ws, size_t ws_size,
                              hipStream_t stream) {
    (void)in_sizes; (void)n_in; (void)out_size; (void)d_ws; (void)ws_size;
    const float* Q = (const float*)d_in[0];
    const float* K = (const float*)d_in[1];
    const float* V = (const float*)d_in[2];
    float* O = (float*)d_out;
    const int nblocks = BB * HN * NWIN;   // 16384 windows
    swin_attn_wmma<<<dim3(nblocks), dim3(128), 0, stream>>>(Q, K, V, O);
}
